// GCSAN_KG_Model_64218351010448
// MI455X (gfx1250) — matrix operand order repaired and checked by reading, not compiled, not run on hardware
//
#include <hip/hip_runtime.h>

// ---------------------------------------------------------------------------
// GCSAN forward for MI455X (gfx1250, wave32).
// All matmuls run on v_wmma_f32_16x16x32_f16 (f16 in, fp32 accumulate).
// Fragments are built from contiguous 16B LDS chunks -> ds_load_b128.
// Model dims: B=512, L=200, D=64, HEADS=2, HD=32, DFF=256, LAYERS=2.
// ---------------------------------------------------------------------------

#define Bc   512
#define Lc   200
#define Dc   64
#define NHc  2
#define HDc  32
#define DFFc 256

typedef __attribute__((ext_vector_type(16))) _Float16 v16h;
typedef __attribute__((ext_vector_type(8)))  _Float16 v8h;
typedef __attribute__((ext_vector_type(4)))  _Float16 v4h;
typedef __attribute__((ext_vector_type(8)))  float    v8f;

#define SHUF16(lo, hi) \
  __builtin_shufflevector(lo, hi, 0, 1, 2, 3, 4, 5, 6, 7, 8, 9, 10, 11, 12, 13, 14, 15)

static __device__ __forceinline__ v8f wmma_f16(v16h a, v16h b, v8f c) {
  return __builtin_amdgcn_wmma_f32_16x16x32_f16(false, a, false, b, (short)0, c,
                                                false, false);
}

static __device__ __forceinline__ v4h cvt4(float x, float y, float z, float w) {
  v4h h;
  h[0] = (_Float16)x; h[1] = (_Float16)y; h[2] = (_Float16)z; h[3] = (_Float16)w;
  return h;
}

// ---------------------------------------------------------------------------
// Fragment layouts (CDNA5 ISA 7.12.2, wave32), with tiles stored K-contiguous:
//  A 16x32 f16 : lane row m = (lane&15)+wave*16, agrp = (lane>>4)*8
//                a = { T[m][agrp..agrp+7], T[m][agrp+16..agrp+23] }  (2x b128)
//  B 32x16 f16 : lane col n = lane&15, bgrp = (lane>>4)*16
//                b = { T[n][agrp..agrp+7], T[n][agrp+16..agrp+23] }                          (2x b128)
//  C/D 16x16 f32: vgpr r -> M = r + (lane>=16 ? 8 : 0), N = lane&15
// ---------------------------------------------------------------------------

// Y[m,n] = act( sum_k X[m*lda+k]*W[n*K+k] + bias[n] ),  Y row-stride = ldy.
// grid.x = M/64, grid.y = N/64, block = 128 (4 waves). K multiple of 32.
__global__ __launch_bounds__(128) void gemm_xwt(
    const float* __restrict__ X, int lda,
    const float* __restrict__ W, const float* __restrict__ bias,
    float* __restrict__ Y, int ldy, int K, int act)
{
  const int m0 = blockIdx.x * 64;
  const int n0 = blockIdx.y * 64;
  __shared__ __align__(16) _Float16 Xs[64][40];   // [m][k]
  __shared__ __align__(16) _Float16 Wt[64][40];   // [n][k]
  const int lane = threadIdx.x & 31;
  const int wave = threadIdx.x >> 5;
  const int mrow = wave * 16 + (lane & 15);
  const int agrp = (lane >> 4) << 3;
  const int bgrp = (lane >> 4) << 4;
  const int nl   = lane & 15;
  v8f acc[4] = {};

  for (int kk = 0; kk < K; kk += 32) {
    __syncthreads();
    for (int t = threadIdx.x; t < 64 * 8; t += 128) {
      int r = t >> 3, c4 = (t & 7) * 4;
      float4 v = *(const float4*)(X + (size_t)(m0 + r) * lda + kk + c4);
      *(v4h*)&Xs[r][c4] = cvt4(v.x, v.y, v.z, v.w);
    }
    for (int t = threadIdx.x; t < 64 * 8; t += 128) {
      int c = t >> 3, r4 = (t & 7) * 4;
      float4 v = *(const float4*)(W + (size_t)(n0 + c) * K + kk + r4);
      *(v4h*)&Wt[c][r4] = cvt4(v.x, v.y, v.z, v.w);
    }
    __syncthreads();
    v8h xlo = *(const v8h*)&Xs[mrow][agrp];
    v8h xhi = *(const v8h*)&Xs[mrow][agrp + 16];
    v16h a  = SHUF16(xlo, xhi);
#pragma unroll
    for (int t = 0; t < 4; ++t) {
      int nc = t * 16 + nl;
      v8h wlo = *(const v8h*)&Wt[nc][agrp];
      v8h whi = *(const v8h*)&Wt[nc][agrp + 16];
      v16h bf = SHUF16(wlo, whi);
      acc[t] = wmma_f16(a, bf, acc[t]);
    }
  }
  const int mbase = m0 + wave * 16 + ((lane >> 4) << 3);
#pragma unroll
  for (int t = 0; t < 4; ++t) {
    int n = n0 + t * 16 + nl;
    float bb = bias ? bias[n] : 0.0f;
#pragma unroll
    for (int r = 0; r < 8; ++r) {
      float v = acc[t][r] + bb;
      if (act) v = fmaxf(v, 0.0f);
      Y[(size_t)(mbase + r) * ldy + n] = v;
    }
  }
}

// Out[b,m,n] = sum_{j<200} A[b, m, colofs+j] * H[b, j, n]   (N = 64)
// A row stride = 2L = 400; K padded 200->224 with zero fill.
// grid = (4, B), block = 128.
__global__ __launch_bounds__(128) void graph_mm(
    const float* __restrict__ Ag, int colofs,
    const float* __restrict__ H,
    float* __restrict__ Out, int ldo)
{
  const int b  = blockIdx.y;
  const int m0 = blockIdx.x * 64;
  __shared__ __align__(16) _Float16 As[64][40];   // [m][k]
  __shared__ __align__(16) _Float16 Ht[64][40];   // [n][k] (transposed H)
  const int lane = threadIdx.x & 31;
  const int wave = threadIdx.x >> 5;
  const int mrow = wave * 16 + (lane & 15);
  const int agrp = (lane >> 4) << 3;
  const int bgrp = (lane >> 4) << 4;
  const int nl   = lane & 15;
  const float* Ab = Ag + (size_t)b * Lc * (2 * Lc) + colofs;
  v8f acc[4] = {};

  for (int kk = 0; kk < 224; kk += 32) {
    __syncthreads();
    for (int t = threadIdx.x; t < 64 * 8; t += 128) {
      int r = t >> 3, c4 = (t & 7) * 4;
      int m = m0 + r, k = kk + c4;
      v4h h4 = {};
      if (m < Lc) {
        if (k + 3 < Lc) {
          float4 v = *(const float4*)(Ab + (size_t)m * (2 * Lc) + k);
          h4 = cvt4(v.x, v.y, v.z, v.w);
        } else {
#pragma unroll
          for (int j = 0; j < 4; ++j)
            h4[j] = (k + j < Lc) ? (_Float16)Ab[(size_t)m * (2 * Lc) + k + j]
                                 : (_Float16)0.0f;
        }
      }
      *(v4h*)&As[r][c4] = h4;
    }
    // transpose-stage H tile: Ht[n][k] = H[b, kk+k, n]
    for (int t = threadIdx.x; t < 32 * 16; t += 128) {
      int k = t >> 4, c4 = (t & 15) * 4;
      float4 v = make_float4(0.f, 0.f, 0.f, 0.f);
      if (kk + k < Lc)
        v = *(const float4*)(H + ((size_t)b * Lc + kk + k) * Dc + c4);
      Ht[c4 + 0][k] = (_Float16)v.x;
      Ht[c4 + 1][k] = (_Float16)v.y;
      Ht[c4 + 2][k] = (_Float16)v.z;
      Ht[c4 + 3][k] = (_Float16)v.w;
    }
    __syncthreads();
    v8h alo = *(const v8h*)&As[mrow][agrp];
    v8h ahi = *(const v8h*)&As[mrow][agrp + 16];
    v16h a  = SHUF16(alo, ahi);
#pragma unroll
    for (int t = 0; t < 4; ++t) {
      int nc = t * 16 + nl;
      v8h blo = *(const v8h*)&Ht[nc][agrp];
      v8h bhi = *(const v8h*)&Ht[nc][agrp + 16];
      v16h bf = SHUF16(blo, bhi);
      acc[t] = wmma_f16(a, bf, acc[t]);
    }
  }
  const int mb = m0 + wave * 16 + ((lane >> 4) << 3);
#pragma unroll
  for (int t = 0; t < 4; ++t) {
    int n = t * 16 + nl;
#pragma unroll
    for (int r = 0; r < 8; ++r) {
      int m = mb + r;
      if (m < Lc) Out[((size_t)b * Lc + m) * ldo + n] = acc[t][r];
    }
  }
}

// Fused attention for one (batch, head, 64 q-rows) strip.
// qkv layout: (B*L, 192) = [q | k | v]; head h owns 32 cols of each third.
// grid = (4, NH, B), block = 128 (4 waves). 1/sqrt(HD) folded into Q.
__global__ __launch_bounds__(128) void attn_kernel(
    const float* __restrict__ qkv, float* __restrict__ out)
{
  const int b  = blockIdx.z;
  const int h  = blockIdx.y;
  const int m0 = blockIdx.x * 64;
  __shared__ __align__(16) _Float16 sc[64][240];  // f16 scores -> probs
  __shared__ __align__(16) char aux[21760];       // qs+Kt, later Vt
  _Float16 (*qs)[40]  = (_Float16(*)[40])aux;             //  5,120 B
  _Float16 (*Kt)[40]  = (_Float16(*)[40])(aux + 5120);    // 16,640 B (208 rows)
  _Float16 (*Vt)[232] = (_Float16(*)[232])aux;            // 14,848 B (phase 2)
  const int lane = threadIdx.x & 31;
  const int wave = threadIdx.x >> 5;
  const int mrow = wave * 16 + (lane & 15);
  const int agrp = (lane >> 4) << 3;
  const int bgrp = (lane >> 4) << 4;
  const int nl   = lane & 15;
  const float scale = 0.17677669529663687f;  // 1/sqrt(32)
  const float NEG = -65504.0f;               // f16 lowest

  // stage Q (scaled) and all K rows (zero-padded to 208)
  for (int t = threadIdx.x; t < 64 * 8; t += 128) {
    int r = t >> 3, c4 = (t & 7) * 4;
    int m = m0 + r;
    float4 v = make_float4(0.f, 0.f, 0.f, 0.f);
    if (m < Lc)
      v = *(const float4*)(qkv + ((size_t)b * Lc + m) * 192 + h * HDc + c4);
    *(v4h*)&qs[r][c4] = cvt4(v.x * scale, v.y * scale, v.z * scale, v.w * scale);
  }
  for (int t = threadIdx.x; t < 208 * 8; t += 128) {
    int r = t >> 3, c4 = (t & 7) * 4;
    float4 v = make_float4(0.f, 0.f, 0.f, 0.f);
    if (r < Lc)
      v = *(const float4*)(qkv + ((size_t)b * Lc + r) * 192 + Dc + h * HDc + c4);
    *(v4h*)&Kt[r][c4] = cvt4(v.x, v.y, v.z, v.w);
  }
  __syncthreads();

  v8h qlo = *(const v8h*)&qs[mrow][agrp];
  v8h qhi = *(const v8h*)&qs[mrow][agrp + 16];
  v16h aq = SHUF16(qlo, qhi);

  // S = (Q/sqrt(HD)) @ K^T : 13 tiles of 16 k-rows, one WMMA each (K=32)
  for (int j = 0; j < 13; ++j) {
    int krow = j * 16 + nl;
    v8h blo = *(const v8h*)&Kt[krow][agrp];
    v8h bhi = *(const v8h*)&Kt[krow][agrp + 16];
    v16h bf = SHUF16(blo, bhi);
    v8f s = {};
    s = wmma_f16(aq, bf, s);
    bool kv = krow < Lc;
    int rb  = wave * 16 + ((lane >> 4) << 3);
#pragma unroll
    for (int r = 0; r < 8; ++r)
      sc[rb + r][krow] = kv ? (_Float16)s[r] : (_Float16)NEG;
  }
  // pad columns 208..239
  for (int t = threadIdx.x; t < 64 * 32; t += 128)
    sc[t >> 5][208 + (t & 31)] = (_Float16)NEG;
  __syncthreads();

  // row softmax (one thread per q-row), f16 storage / f32 math
  if (threadIdx.x < 64) {
    _Float16* rp = sc[threadIdx.x];
    float mx = NEG;
    for (int c = 0; c < 240; ++c) mx = fmaxf(mx, (float)rp[c]);
    float s = 0.0f;
    for (int c = 0; c < 240; ++c) {
      float e = __expf((float)rp[c] - mx);
      s += e;
      rp[c] = (_Float16)e;
    }
    float inv = 1.0f / s;
    for (int c = 0; c < 240; ++c) rp[c] = (_Float16)((float)rp[c] * inv);
  }
  __syncthreads();

  // stage V transposed: Vt[n][kr] = V[kr][n]  (kr zero-padded to 224)
  for (int t = threadIdx.x; t < 224 * 8; t += 128) {
    int r = t >> 3, c4 = (t & 7) * 4;
    float4 v = make_float4(0.f, 0.f, 0.f, 0.f);
    if (r < Lc)
      v = *(const float4*)(qkv + ((size_t)b * Lc + r) * 192 + 2 * Dc + h * HDc + c4);
    Vt[c4 + 0][r] = (_Float16)v.x;
    Vt[c4 + 1][r] = (_Float16)v.y;
    Vt[c4 + 2][r] = (_Float16)v.z;
    Vt[c4 + 3][r] = (_Float16)v.w;
  }
  __syncthreads();

  // O = P @ V : K padded to 224 (pad probs are exactly 0)
  v8f ao[2] = {};
  for (int kk = 0; kk < 224; kk += 32) {
    v8h plo = *(const v8h*)&sc[mrow][kk + agrp];
    v8h phi = *(const v8h*)&sc[mrow][kk + agrp + 16];
    v16h ap = SHUF16(plo, phi);
#pragma unroll
    for (int t = 0; t < 2; ++t) {
      int n = t * 16 + nl;
      v8h vlo = *(const v8h*)&Vt[n][kk + agrp];
      v8h vhi = *(const v8h*)&Vt[n][kk + agrp + 16];
      v16h bf = SHUF16(vlo, vhi);
      ao[t] = wmma_f16(ap, bf, ao[t]);
    }
  }
  const int mb = m0 + wave * 16 + ((lane >> 4) << 3);
#pragma unroll
  for (int t = 0; t < 2; ++t) {
    int n = t * 16 + nl;
#pragma unroll
    for (int r = 0; r < 8; ++r) {
      int m = mb + r;
      if (m < Lc)
        out[((size_t)b * Lc + m) * Dc + h * HDc + n] = ao[t][r];
    }
  }
}

// hidden <- GRU(gi, gh, hidden)
__global__ void gru_kernel(const float* __restrict__ gi,
                           const float* __restrict__ gh,
                           const float* __restrict__ h,
                           float* __restrict__ hnew, int total)
{
  int idx = blockIdx.x * blockDim.x + threadIdx.x;
  if (idx >= total) return;
  int row = idx >> 6, d = idx & 63;
  const float* gir = gi + (size_t)row * 192;
  const float* ghr = gh + (size_t)row * 192;
  float rg = 1.0f / (1.0f + __expf(-(gir[d] + ghr[d])));
  float ig = 1.0f / (1.0f + __expf(-(gir[64 + d] + ghr[64 + d])));
  float ng = tanhf(gir[128 + d] + rg * ghr[128 + d]);
  hnew[idx] = (1.0f - ig) * h[idx] + ig * ng;
}

// y = LayerNorm(x (+ res)) * g + b ; one wave per 64-wide row.
__global__ void ln_kernel(const float* __restrict__ x,
                          const float* __restrict__ res,
                          const float* __restrict__ g,
                          const float* __restrict__ bta,
                          float* __restrict__ y, int rows)
{
  int wid  = (int)((blockIdx.x * (size_t)blockDim.x + threadIdx.x) >> 5);
  int lane = threadIdx.x & 31;
  if (wid >= rows) return;
  size_t base = (size_t)wid * 64;
  float v0 = x[base + lane], v1 = x[base + lane + 32];
  if (res) { v0 += res[base + lane]; v1 += res[base + lane + 32]; }
  float s = v0 + v1;
  for (int o = 16; o > 0; o >>= 1) s += __shfl_xor(s, o, 32);
  float mean = s * (1.0f / 64.0f);
  float d0 = v0 - mean, d1 = v1 - mean;
  float q = d0 * d0 + d1 * d1;
  for (int o = 16; o > 0; o >>= 1) q += __shfl_xor(q, o, 32);
  float rstd = rsqrtf(q * (1.0f / 64.0f) + 1e-5f);
  y[base + lane]      = d0 * rstd * g[lane]      + bta[lane];
  y[base + lane + 32] = d1 * rstd * g[lane + 32] + bta[lane + 32];
}

// hidden[row,:] = item_emb[items[row],:]   (float4 vectorized)
__global__ void embed_kernel(const float* __restrict__ emb,
                             const int* __restrict__ items,
                             float* __restrict__ out, int nrows)
{
  int t = blockIdx.x * blockDim.x + threadIdx.x;
  if (t >= nrows * 16) return;
  int row = t >> 4, c = (t & 15) * 4;
  int it = items[row];
  *(float4*)(out + (size_t)row * 64 + c) =
      *(const float4*)(emb + (size_t)it * 64 + c);
}

// x[b,l,:] = h[b, alias[b,l], :]
__global__ void alias_kernel(const float* __restrict__ h,
                             const int* __restrict__ alias,
                             float* __restrict__ out, int nrows)
{
  int t = blockIdx.x * blockDim.x + threadIdx.x;
  if (t >= nrows * 16) return;
  int row = t >> 4, c = (t & 15) * 4;
  int b = row / Lc;
  int a = alias[row];
  *(float4*)(out + (size_t)row * 64 + c) =
      *(const float4*)(h + ((size_t)b * Lc + a) * 64 + c);
}

// ---------------------------------------------------------------------------

extern "C" void kernel_launch(void* const* d_in, const int* in_sizes, int n_in,
                              void* d_out, int out_size, void* d_ws, size_t ws_size,
                              hipStream_t stream)
{
  (void)in_sizes; (void)out_size;
  // Input order: A, alias_inputs, items, then params flattened as a JAX
  // pytree (dict keys sorted; ASCII order, capitals < '_' < lowercase):
  //  3:W_in 4:W_out 5:Wfc 6:bF 7:b_hh 8:b_ih 9:b_in 10:b_out 11:bfc 12:gF
  //  13:item_emb 14..25:layer0{W1,W2,Wo,Wqkv,b1,b2,be1,be2,bo,bqkv,g1,g2}
  //  26..37:layer1{...} 38:w_hh 39:w_ih
  if (n_in < 40) return;
  const float* Ain   = (const float*)d_in[0];
  const int*   alias = (const int*)d_in[1];
  const int*   items = (const int*)d_in[2];
  const float* W_in  = (const float*)d_in[3];
  const float* W_out = (const float*)d_in[4];
  const float* Wfc   = (const float*)d_in[5];
  const float* bF    = (const float*)d_in[6];
  const float* b_hh  = (const float*)d_in[7];
  const float* b_ih  = (const float*)d_in[8];
  const float* b_in  = (const float*)d_in[9];
  const float* b_out = (const float*)d_in[10];
  const float* bfc   = (const float*)d_in[11];
  const float* gF    = (const float*)d_in[12];
  const float* emb   = (const float*)d_in[13];
  const float* w_hh  = (const float*)d_in[38];
  const float* w_ih  = (const float*)d_in[39];
  float* out = (float*)d_out;

  const size_t NBL = (size_t)Bc * Lc;       // 102400
  // Workspace layout (floats), with region reuse:
  //  [0)        hidden   NBL*64   (later: x1)
  //  [1*NBL*64) h_in     NBL*64   (later: hnew, attn_out)
  //  [2*NBL*64) h_out    NBL*64   (later: x)
  //  [3*NBL*64) inputs   NBL*128  (later: proj, ff2)
  //  [5*NBL*64) gi       NBL*192  (later: qkv; with gh region: ff1 NBL*256)
  //  [8*NBL*64) gh       NBL*192
  const size_t need = 11 * NBL * 64 * sizeof(float);
  if (ws_size < need) return;
  float* ws     = (float*)d_ws;
  float* hidden = ws;
  float* hin    = ws + 1 * NBL * 64;
  float* hout   = ws + 2 * NBL * 64;
  float* inputs = ws + 3 * NBL * 64;
  float* gi     = ws + 5 * NBL * 64;
  float* gh     = ws + 8 * NBL * 64;
  float* hnew = hin;
  float* x    = hout;
  float* qkvb = gi;
  float* attn = hin;
  float* proj = inputs;
  float* x1   = hidden;
  float* ff1  = gi;
  float* ff2  = inputs;

  const int MT = (int)(NBL / 64);           // 1600 M-tiles
  dim3 blk(128);

  // 1) embedding gather
  embed_kernel<<<(int)((NBL * 16 + 255) / 256), 256, 0, stream>>>(emb, items, hidden, (int)NBL);

  // 2) GNN cell
  gemm_xwt<<<dim3(MT, 1), blk, 0, stream>>>(hidden, 64, W_in,  b_in,  hin,  64, 64, 0);
  gemm_xwt<<<dim3(MT, 1), blk, 0, stream>>>(hidden, 64, W_out, b_out, hout, 64, 64, 0);
  graph_mm<<<dim3(4, Bc), blk, 0, stream>>>(Ain, 0,  hin,  inputs,      128);
  graph_mm<<<dim3(4, Bc), blk, 0, stream>>>(Ain, Lc, hout, inputs + 64, 128);
  gemm_xwt<<<dim3(MT, 3), blk, 0, stream>>>(inputs, 128, w_ih, b_ih, gi, 192, 128, 0);
  gemm_xwt<<<dim3(MT, 3), blk, 0, stream>>>(hidden,  64, w_hh, b_hh, gh, 192,  64, 0);
  gru_kernel<<<(int)((NBL * 64 + 255) / 256), 256, 0, stream>>>(gi, gh, hidden, hnew, (int)(NBL * 64));

  // 3) alias gather -> x
  alias_kernel<<<(int)((NBL * 16 + 255) / 256), 256, 0, stream>>>(hnew, alias, x, (int)NBL);

  // 4) transformer encoder layers
  for (int l = 0; l < 2; ++l) {
    int base = 14 + l * 12;
    const float* W1   = (const float*)d_in[base + 0];
    const float* W2   = (const float*)d_in[base + 1];
    const float* Wo   = (const float*)d_in[base + 2];
    const float* Wqkv = (const float*)d_in[base + 3];
    const float* b1   = (const float*)d_in[base + 4];
    const float* b2   = (const float*)d_in[base + 5];
    const float* be1  = (const float*)d_in[base + 6];
    const float* be2  = (const float*)d_in[base + 7];
    const float* bo   = (const float*)d_in[base + 8];
    const float* bqkv = (const float*)d_in[base + 9];
    const float* g1   = (const float*)d_in[base + 10];
    const float* g2   = (const float*)d_in[base + 11];

    gemm_xwt<<<dim3(MT, 3), blk, 0, stream>>>(x, 64, Wqkv, bqkv, qkvb, 192, 64, 0);
    attn_kernel<<<dim3(4, NHc, Bc), blk, 0, stream>>>(qkvb, attn);
    gemm_xwt<<<dim3(MT, 1), blk, 0, stream>>>(attn, 64, Wo, bo, proj, 64, 64, 0);
    ln_kernel<<<(int)(NBL / 8), 256, 0, stream>>>(x, proj, g1, be1, x1, (int)NBL);
    gemm_xwt<<<dim3(MT, 4), blk, 0, stream>>>(x1, 64, W1, b1, ff1, 256, 64, 1);   // ReLU
    gemm_xwt<<<dim3(MT, 1), blk, 0, stream>>>(ff1, 256, W2, b2, ff2, 64, 256, 0);
    ln_kernel<<<(int)(NBL / 8), 256, 0, stream>>>(x1, ff2, g2, be2, x, (int)NBL);
  }

  // 5) final LN, then last-token fc:  out = xF[:, L-1, :] @ Wfc^T + bfc
  ln_kernel<<<(int)(NBL / 8), 256, 0, stream>>>(x, nullptr, gF, bF, hidden, (int)NBL);
  gemm_xwt<<<dim3(Bc / 64, 1), blk, 0, stream>>>(hidden + (size_t)(Lc - 1) * 64,
                                                 Lc * 64, Wfc, bfc, out, 64, 64, 0);
}
